// SDPAttention_44873818308884
// MI455X (gfx1250) — matrix operand order repaired and checked by reading, not compiled, not run on hardware
//
#include <hip/hip_runtime.h>
#include <hip/hip_bf16.h>
#include <math.h>

// ---------------------------------------------------------------------------
// Problem constants (B, C, H, W) = (4, 128, 64, 64), L = H*W = 4096
// ---------------------------------------------------------------------------
constexpr int BB = 4;
constexpr int CC = 128;
constexpr int LL = 4096;           // 64*64
constexpr int KT = 64;             // key tile per iteration
// (1/sqrt(C)) * log2(e): scores scaled into log2 domain, exp2f per element
constexpr float SC2 = 0.08838834764831845f * 1.4426950408889634f;

typedef __attribute__((ext_vector_type(16))) __bf16 v16bf;
typedef __attribute__((ext_vector_type(8)))  float  v8f;

union V16BF {
    v16bf v;
    uint4 q[2];
    __bf16 h[16];
};

// Async memory -> LDS copy, 16B per lane, tracked by ASYNCcnt.
// offset applies to BOTH the global address and the LDS address (ISA 10.x).
#define ASYNC_B128(ldsoff, gaddr, OFF)                                         \
    asm volatile("global_load_async_to_lds_b128 %0, %1, off offset:" #OFF      \
                 :: "v"(ldsoff), "v"(gaddr) : "memory")

// ---------------------------------------------------------------------------
// Kernel 0: x[b,c,l] (f32) -> Q[b,l,c] (bf16).  Coalesced writes.
// ---------------------------------------------------------------------------
__global__ __launch_bounds__(256) void k_transpose_bf16(const float* __restrict__ x,
                                                        __bf16* __restrict__ Q) {
    size_t idx = (size_t)blockIdx.x * 256 + threadIdx.x;   // over B*L*C
    int c = (int)(idx & (CC - 1));
    size_t bl = idx >> 7;            // /128
    int l = (int)(bl & (LL - 1));
    int b = (int)(bl >> 12);         // /4096
    Q[idx] = (__bf16)x[((size_t)b * CC + c) * LL + l];
}

// ---------------------------------------------------------------------------
// Kernel 1: V = W_kv * X + bias  (per batch 128x4096 GEMM, K=128)
// One wave per 16x16 output tile.  Writes V in both [l,c] and [c,l] layouts.
// ---------------------------------------------------------------------------
__global__ __launch_bounds__(32) void k_kv_gemm(const float* __restrict__ Wkv,
                                                const float* __restrict__ bias,
                                                const __bf16* __restrict__ Q,
                                                __bf16* __restrict__ Vlc,
                                                __bf16* __restrict__ Vcl) {
    const int b  = blockIdx.z;
    const int c0 = blockIdx.y * 16;   // output-channel tile
    const int l0 = blockIdx.x * 16;   // spatial tile
    const int lane = threadIdx.x & 31;
    const int n = lane & 15;          // N index (and A-row index m)
    const int g = lane >> 4;          // lane half

    v8f acc = {0.f, 0.f, 0.f, 0.f, 0.f, 0.f, 0.f, 0.f};

#pragma unroll
    for (int f = 0; f < 4; ++f) {
        // A fragment: W_kv rows c0+m, K = 32f..32f+31 (convert f32->bf16)
        V16BF a;
        const float* wa = Wkv + (c0 + n) * CC + f * 32 + g * 16;
#pragma unroll
        for (int e = 0; e < 8; ++e) {
            a.h[e]     = (__bf16)wa[e];        // K = 16h + e
            a.h[e + 8] = (__bf16)wa[8 + e];    // K = 16h + 8 + e
        }
        // B fragment: Q[b, l0+n, 32f + 16g + e]  (16 contiguous bf16)
        V16BF bf;
        const __bf16* qb = Q + ((size_t)b * LL + l0 + n) * CC + f * 32 + g * 16;
        bf.q[0] = ((const uint4*)qb)[0];
        bf.q[1] = ((const uint4*)qb)[1];
        acc = __builtin_amdgcn_wmma_f32_16x16x32_bf16(false, a.v, false, bf.v,
                                                      (short)0, acc, false, false);
    }

    // Bias + dual-layout store.  D layout: VGPR r holds M = r + 8g, lane n = N.
    union { uint4 u; __bf16 h[8]; } pk;
#pragma unroll
    for (int r = 0; r < 8; ++r) {
        float v = acc[r] + bias[c0 + 8 * g + r];
        pk.h[r] = (__bf16)v;
        Vcl[((size_t)b * CC + c0 + 8 * g + r) * LL + l0 + n] = (__bf16)v;
    }
    *(uint4*)(Vlc + ((size_t)b * LL + l0 + n) * CC + c0 + 8 * g) = pk.u;
}

// ---------------------------------------------------------------------------
// Kernel 2: flash attention.  Block = 128 threads (4 waves), each wave owns
// 16 query rows; 64-key tiles, double-buffered async LDS staging.
// ---------------------------------------------------------------------------
__global__ __launch_bounds__(128) void k_attention(const __bf16* __restrict__ Q,
                                                   const __bf16* __restrict__ Klc,
                                                   const __bf16* __restrict__ Vcl,
                                                   float* __restrict__ out) {
    const int b    = blockIdx.y;
    const int w    = threadIdx.x >> 5;   // wave 0..3
    const int lane = threadIdx.x & 31;
    const int n = lane & 15;             // N / A-row index
    const int g = lane >> 4;             // lane half
    const int qrow0 = blockIdx.x * 64 + w * 16;
    const int t = threadIdx.x;

    __shared__ __bf16 Klds[2][KT][CC];   // [buf][key][channel]  32 KB
    __shared__ __bf16 Vlds[2][CC][KT];   // [buf][channel][key]  32 KB
    __shared__ __bf16 Plds[4][16][KT];   // per-wave P tile       8 KB

    // ---- async staging addresses (16B per lane per op) --------------------
    // K tile: 64 rows x 256B; thread covers row t>>1, 128B half (t&1).
    // V tile: 128 rows x 128B; thread covers row t.
    uint64_t gk = (uint64_t)(Klc + ((size_t)b * LL + (t >> 1)) * CC + (t & 1) * 64);
    uint64_t gv = (uint64_t)(Vcl + ((size_t)b * CC + t) * LL);
    unsigned lk[2], lv[2];
    lk[0] = (unsigned)(uintptr_t)&Klds[0][t >> 1][(t & 1) * 64];
    lk[1] = (unsigned)(uintptr_t)&Klds[1][t >> 1][(t & 1) * 64];
    lv[0] = (unsigned)(uintptr_t)&Vlds[0][t][0];
    lv[1] = (unsigned)(uintptr_t)&Vlds[1][t][0];

    auto issue_tile = [&](int p) {
        const unsigned a = lk[p];
        ASYNC_B128(a, gk, 0);   ASYNC_B128(a, gk, 16);
        ASYNC_B128(a, gk, 32);  ASYNC_B128(a, gk, 48);
        ASYNC_B128(a, gk, 64);  ASYNC_B128(a, gk, 80);
        ASYNC_B128(a, gk, 96);  ASYNC_B128(a, gk, 112);
        const unsigned c = lv[p];
        ASYNC_B128(c, gv, 0);   ASYNC_B128(c, gv, 16);
        ASYNC_B128(c, gv, 32);  ASYNC_B128(c, gv, 48);
        ASYNC_B128(c, gv, 64);  ASYNC_B128(c, gv, 80);
        ASYNC_B128(c, gv, 96);  ASYNC_B128(c, gv, 112);
        gk += (uint64_t)KT * CC * sizeof(__bf16);   // next 64 key rows
        gv += (uint64_t)KT * sizeof(__bf16);        // next 64 key cols
    };

    // ---- Q fragments (A-layout), resident in registers for the whole loop -
    V16BF aq[4];
    {
        const __bf16* qbase = Q + ((size_t)b * LL + qrow0 + n) * CC;
#pragma unroll
        for (int f = 0; f < 4; ++f) {
            const __bf16* p = qbase + f * 32 + g * 16;
            aq[f].q[0] = ((const uint4*)p)[0];          // K = 16h+e
            aq[f].q[1] = *(const uint4*)(p + 8);        // K = 16h+8+e
        }
    }

    v8f o[8];
    float m_r[8], l_r[8];
#pragma unroll
    for (int j = 0; j < 8; ++j) o[j] = (v8f){0.f, 0.f, 0.f, 0.f, 0.f, 0.f, 0.f, 0.f};
#pragma unroll
    for (int r = 0; r < 8; ++r) { m_r[r] = -INFINITY; l_r[r] = 0.f; }

    issue_tile(0);   // prefetch first tile

    constexpr int NIT = LL / KT;   // 64 iterations
    for (int it = 0; it < NIT; ++it) {
        const int p = it & 1;
        // drain our async fills of buffer p, then rendezvous: after the
        // barrier every wave's fills landed AND every wave is done reading
        // buffer p^1 from the previous iteration.
        asm volatile("s_wait_asynccnt 0" ::: "memory");
        __syncthreads();
        if (it + 1 < NIT) issue_tile(p ^ 1);   // overlaps with compute below

        // ---- S = Q K^T : 16 rows x 64 keys (four 16x16 tiles) ----
        v8f s[4];
#pragma unroll
        for (int k = 0; k < 4; ++k) s[k] = (v8f){0.f, 0.f, 0.f, 0.f, 0.f, 0.f, 0.f, 0.f};
#pragma unroll
        for (int f = 0; f < 4; ++f) {
#pragma unroll
            for (int k = 0; k < 4; ++k) {
                V16BF kb;
                const __bf16* kp = &Klds[p][16 * k + n][f * 32 + g * 16];
                kb.q[0] = ((const uint4*)kp)[0];
                kb.q[1] = ((const uint4*)kp)[1];
                s[k] = __builtin_amdgcn_wmma_f32_16x16x32_bf16(false, aq[f].v, false, kb.v,
                                                               (short)0, s[k], false, false);
            }
        }

        // ---- online softmax in log2 domain (row = r + 8g over 16 lanes) ---
#pragma unroll
        for (int r = 0; r < 8; ++r) {
            const float v0 = s[0][r] * SC2;
            const float v1 = s[1][r] * SC2;
            const float v2 = s[2][r] * SC2;
            const float v3 = s[3][r] * SC2;
            float tmax = fmaxf(fmaxf(v0, v1), fmaxf(v2, v3));
#pragma unroll
            for (int off = 1; off < 16; off <<= 1)
                tmax = fmaxf(tmax, __shfl_xor(tmax, off, 32));
            const float mn = fmaxf(m_r[r], tmax);
            const float sc = exp2f(m_r[r] - mn);
            const float p0 = exp2f(v0 - mn);
            const float p1 = exp2f(v1 - mn);
            const float p2 = exp2f(v2 - mn);
            const float p3 = exp2f(v3 - mn);
            float rs = (p0 + p1) + (p2 + p3);
#pragma unroll
            for (int off = 1; off < 16; off <<= 1)
                rs += __shfl_xor(rs, off, 32);
            l_r[r] = l_r[r] * sc + rs;
            m_r[r] = mn;
#pragma unroll
            for (int j = 0; j < 8; ++j) o[j][r] *= sc;
            // stash P (C/D layout -> row-major bf16 tile in per-wave LDS)
            __bf16* prow = &Plds[w][r + 8 * g][0];
            prow[n]      = (__bf16)p0;
            prow[16 + n] = (__bf16)p1;
            prow[32 + n] = (__bf16)p2;
            prow[48 + n] = (__bf16)p3;
        }

        // DS ops are in-order per wave; drain stores before cross-lane reads.
        asm volatile("s_wait_dscnt 0" ::: "memory");

        // ---- P as two A-fragments (keys 0..31, 32..63), O += P * V --------
        V16BF ap[2];
#pragma unroll
        for (int tt = 0; tt < 2; ++tt) {
            const __bf16* pp = &Plds[w][n][32 * tt + g * 16];
            ap[tt].q[0] = ((const uint4*)pp)[0];        // K = 16h+e
            ap[tt].q[1] = *(const uint4*)(pp + 8);      // K = 16h+8+e
        }
#pragma unroll
        for (int j = 0; j < 8; ++j) {
#pragma unroll
            for (int tt = 0; tt < 2; ++tt) {
                V16BF vb;
                const __bf16* vp = &Vlds[p][16 * j + n][32 * tt + g * 16];
                vb.q[0] = ((const uint4*)vp)[0];
                vb.q[1] = ((const uint4*)vp)[1];
                o[j] = __builtin_amdgcn_wmma_f32_16x16x32_bf16(false, ap[tt].v, false, vb.v,
                                                               (short)0, o[j], false, false);
            }
        }
    }

    // ---- epilogue: divide by row sums, store f32 in [b, c, l] order -------
#pragma unroll
    for (int j = 0; j < 8; ++j) {
        union { v8f v; float4 f4[2]; } ov;
#pragma unroll
        for (int r = 0; r < 8; ++r) ov.v[r] = o[j][r] / l_r[r];
        float* dst = out + ((size_t)b * CC + 16 * j + n) * LL + qrow0 + 8 * g;
        *(float4*)dst       = ov.f4[0];
        *(float4*)(dst + 4) = ov.f4[1];
    }
}

// ---------------------------------------------------------------------------
// Launch
// ---------------------------------------------------------------------------
extern "C" void kernel_launch(void* const* d_in, const int* in_sizes, int n_in,
                              void* d_out, int out_size, void* d_ws, size_t ws_size,
                              hipStream_t stream) {
    const float* x    = (const float*)d_in[0];   // [B, C, H, W]
    const float* Wkv  = (const float*)d_in[1];   // [C, C]
    const float* bkv  = (const float*)d_in[2];   // [C]
    float* out = (float*)d_out;                  // [B, C, H, W]

    const size_t tile = (size_t)BB * LL * CC;    // 2M elements
    __bf16* Q   = (__bf16*)d_ws;                 // [B][L][C] bf16
    __bf16* Vlc = Q + tile;                      // [B][L][C] bf16
    __bf16* Vcl = Vlc + tile;                    // [B][C][L] bf16

    // 0) x -> Q (bf16, [l, c] layout)
    k_transpose_bf16<<<(int)(tile / 256), 256, 0, stream>>>(x, Q);

    // 1) V = W_kv * X + bias (both layouts)
    k_kv_gemm<<<dim3(LL / 16, CC / 16, BB), 32, 0, stream>>>(Wkv, bkv, Q, Vlc, Vcl);

    // 2) flash attention
    k_attention<<<dim3(LL / 64, BB), 128, 0, stream>>>(Q, Vlc, Vcl, out);
}
